// CodeWiseAttention_18124761989378
// MI455X (gfx1250) — matrix operand order repaired and checked by reading, not compiled, not run on hardware
//
#include <hip/hip_runtime.h>

typedef _Float16 v16h __attribute__((ext_vector_type(16)));
typedef _Float16 v8h  __attribute__((ext_vector_type(8)));
typedef float    v8f  __attribute__((ext_vector_type(8)));

#define BATCH 8
#define LSEQ  2500
#define EDIM  100
#define NLAB  8922
#define EPAD  128
#define LBLK  32
#define NWAVE 8
#define NTHR  (NWAVE * 32)
#define NQT   ((NLAB + 15) / 16)   /* 558 query tiles of 16 */

__global__ __launch_bounds__(NTHR)
void cwa_flash_f16wmma(const float* __restrict__ x,
                       const float* __restrict__ label,
                       float* __restrict__ out)
{
  // LDS: x block in two layouts (shared by all waves), per-wave Q tile and P tile
  __shared__ __align__(64) _Float16 x_rm[LBLK][EPAD];        // [key][E]   8 KB
  __shared__ __align__(64) _Float16 x_cm[EPAD][LBLK];        // [E][key]   8 KB
  __shared__ __align__(64) _Float16 q_sm[NWAVE][16][EPAD];   //           32 KB
  __shared__ __align__(64) _Float16 p_sm[NWAVE][16][LBLK];   //            8 KB

  const int tid  = threadIdx.x;
  const int w    = tid >> 5;
  const int lane = tid & 31;
  const int hi   = lane >> 4;      // half-wave select
  const int lo   = lane & 15;
  const int b    = blockIdx.y;
  const int qt   = blockIdx.x * NWAVE + w;
  const bool qok = (qt < NQT);
  const int n0   = (qok ? qt : NQT - 1) * 16;

  // ---- one-time zero fill of E-padding columns (persist across L blocks) ----
  for (int i = tid; i < LBLK * (EPAD - EDIM); i += NTHR) {
    int r = i / (EPAD - EDIM);
    int c = EDIM + (i - r * (EPAD - EDIM));
    x_rm[r][c] = (_Float16)0.f;
    x_cm[c][r] = (_Float16)0.f;
  }
  // ---- per-wave Q tile -> LDS as f16, pre-scaled by log2(e), zero padded ----
  // softmax(s) == softmax_base2(s * log2e): fold the exp scale into Q once.
  for (int i = lane; i < 16 * EDIM; i += 32) {
    int r = i / EDIM;
    int c = i - r * EDIM;
    int q = n0 + r; if (q >= NLAB) q = NLAB - 1;
    q_sm[w][r][c] = (_Float16)(label[(size_t)q * EDIM + c] * 1.44269504088896340736f);
  }
  for (int i = lane; i < 16 * (EPAD - EDIM); i += 32) {
    int r = i / (EPAD - EDIM);
    int c = EDIM + (i - r * (EPAD - EDIM));
    q_sm[w][r][c] = (_Float16)0.f;
  }

  // ---- register-staged x loader: 8 threads per row, cols c0 + 8j ----
  const int lrow = tid >> 3;       // 0..31 (key row within block)
  const int c0   = tid & 7;        // column phase
  float rx[13];
  const float* xbase = x + (size_t)b * LSEQ * EDIM;

  auto fetch = [&](int l0) {
    int gl = l0 + lrow; if (gl > LSEQ - 1) gl = LSEQ - 1;
    const float* gp = xbase + (size_t)gl * EDIM + c0;
    #pragma unroll
    for (int j = 0; j < 12; ++j) rx[j] = gp[8 * j];
    if (c0 < 4) rx[12] = gp[96];          // cols 96..99
  };
  auto stage = [&]() {
    #pragma unroll
    for (int j = 0; j < 12; ++j) {
      _Float16 h = (_Float16)rx[j];
      int c = c0 + 8 * j;
      x_rm[lrow][c] = h;
      x_cm[c][lrow] = h;
    }
    if (c0 < 4) {
      _Float16 h = (_Float16)rx[12];
      x_rm[lrow][96 + c0] = h;
      x_cm[96 + c0][lrow] = h;
    }
  };

  fetch(0);                 // prime the pipeline
  __syncthreads();          // q_sm + padding ready

  // ---- Q fragments in WMMA B-operand layout (lane=query col, elems=K chunk) ----
  v16h qf[4];
  #pragma unroll
  for (int ec = 0; ec < 4; ++ec)
    qf[ec] = __builtin_shufflevector(*(const v8h*)&q_sm[w][lo][ec * 32 + 8 * hi], *(const v8h*)&q_sm[w][lo][ec * 32 + 8 * hi + 16], 0,1,2,3,4,5,6,7,8,9,10,11,12,13,14,15);

  // ---- flash-attention running state (base-2 exponent domain) ----
  float mrun = -__builtin_inff();
  float lrun = 0.f;
  v8f O[8];
  #pragma unroll
  for (int c = 0; c < 8; ++c) {
    #pragma unroll
    for (int r = 0; r < 8; ++r) O[c][r] = 0.f;
  }

  const int nblk = (LSEQ + LBLK - 1) / LBLK;   // 79
  for (int blk = 0; blk < nblk; ++blk) {
    const int l0 = blk * LBLK;
    __syncthreads();    // previous block's LDS reads done before overwrite
    stage();            // registers -> both LDS layouts (f16)
    __syncthreads();    // x tiles ready
    if (blk + 1 < nblk) fetch(l0 + LBLK);   // overlap next global load w/ compute

    // ---- preload ALL 8 A-fragments (both 16-key tiles) before the WMMA chain
    //      so LDS latency is pipelined instead of s_wait_dscnt 0 per WMMA ----
    v16h afr[8];
    #pragma unroll
    for (int t = 0; t < 2; ++t) {
      #pragma unroll
      for (int ec = 0; ec < 4; ++ec) {
        const _Float16* pa = &x_rm[t * 16 + lo][ec * 32 + 8 * hi];
        v8h alo = *(const v8h*)pa;
        v8h ahi = *(const v8h*)(pa + 16);
        afr[t * 4 + ec] = __builtin_shufflevector(alo, ahi,
                            0,1,2,3,4,5,6,7,8,9,10,11,12,13,14,15);
      }
    }

    // ---- S^T = K * Q^T : two 16-key tiles, each 4 WMMAs over padded E ----
    v8f st[2];
    #pragma unroll
    for (int t = 0; t < 2; ++t) {
      v8f acc;
      #pragma unroll
      for (int r = 0; r < 8; ++r) acc[r] = 0.f;
      #pragma unroll
      for (int ec = 0; ec < 4; ++ec)
        acc = __builtin_amdgcn_wmma_f32_16x16x32_f16(
                  false, afr[t * 4 + ec], false, qf[ec], (short)0, acc, false, false);
      st[t] = acc;
    }

    // mask keys beyond sequence end -- only the final block has any (uniform branch)
    if (blk == nblk - 1) {
      #pragma unroll
      for (int t = 0; t < 2; ++t) {
        #pragma unroll
        for (int r = 0; r < 8; ++r)
          if (l0 + t * 16 + 8 * hi + r >= LSEQ) st[t][r] = -__builtin_inff();
      }
    }

    // ---- online softmax stats; lanes (lo, lo+16) jointly own query column lo ----
    float mloc = -__builtin_inff();
    #pragma unroll
    for (int t = 0; t < 2; ++t)
      #pragma unroll
      for (int r = 0; r < 8; ++r) mloc = fmaxf(mloc, st[t][r]);
    mloc = fmaxf(mloc, __shfl_xor(mloc, 16, 32));
    float mnew = fmaxf(mrun, mloc);
    float scal = __builtin_exp2f(mrun - mnew);   // exp2(-inf) = 0 on first block
    mrun = mnew;

    float ssum = 0.f;
    _Float16 pv16[2][8];
    #pragma unroll
    for (int t = 0; t < 2; ++t) {
      #pragma unroll
      for (int r = 0; r < 8; ++r) {
        float p = __builtin_exp2f(st[t][r] - mnew);  // masked -> 0
        ssum += p;
        pv16[t][r] = (_Float16)p;
      }
    }
    ssum += __shfl_xor(ssum, 16, 32);
    lrun = lrun * scal + ssum;

    // ---- write P (transposed back to [q][key]) into per-wave LDS ----
    #pragma unroll
    for (int t = 0; t < 2; ++t) {
      v8h pk;
      #pragma unroll
      for (int r = 0; r < 8; ++r) pk[r] = pv16[t][r];
      *(v8h*)&p_sm[w][lo][t * 16 + 8 * hi] = pk;
    }
    __syncthreads();   // P visible (dscnt flushed), all waves aligned

    // ---- rescale O by exp(m_old - m_new), broadcast per output row ----
    float rs[8];
    #pragma unroll
    for (int r = 0; r < 8; ++r) rs[r] = __shfl(scal, r + 8 * hi, 32);
    #pragma unroll
    for (int c = 0; c < 8; ++c) {
      #pragma unroll
      for (int r = 0; r < 8; ++r) O[c][r] *= rs[r];
    }

    // ---- O += P * V : preload P fragment + all 8 V fragments, then chain ----
    const _Float16* pp = &p_sm[w][lo][8 * hi];
    v8h plo = *(const v8h*)pp;
    v8h phi = *(const v8h*)(pp + 16);
    v16h pa = __builtin_shufflevector(plo, phi,
                0,1,2,3,4,5,6,7,8,9,10,11,12,13,14,15);
    v16h bfr[8];
    #pragma unroll
    for (int c = 0; c < 8; ++c)
      bfr[c] = __builtin_shufflevector(*(const v8h*)&x_cm[c * 16 + lo][8 * hi], *(const v8h*)&x_cm[c * 16 + lo][8 * hi + 16], 0,1,2,3,4,5,6,7,8,9,10,11,12,13,14,15);
    #pragma unroll
    for (int c = 0; c < 8; ++c)
      O[c] = __builtin_amdgcn_wmma_f32_16x16x32_f16(
                 false, pa, false, bfr[c], (short)0, O[c], false, false);
  }

  // ---- finalize: divide by row sums, store guarded ----
  float linv[8];
  #pragma unroll
  for (int r = 0; r < 8; ++r) {
    float lq = __shfl(lrun, r + 8 * hi, 32);
    linv[r] = 1.f / lq;
  }
  if (qok) {
    #pragma unroll
    for (int c = 0; c < 8; ++c) {
      int e = c * 16 + lo;
      if (e < EDIM) {
        #pragma unroll
        for (int r = 0; r < 8; ++r) {
          int q = n0 + r + 8 * hi;
          if (q < NLAB)
            out[((size_t)b * NLAB + q) * EDIM + e] = O[c][r] * linv[r];
        }
      }
    }
  }
}

extern "C" void kernel_launch(void* const* d_in, const int* in_sizes, int n_in,
                              void* d_out, int out_size, void* d_ws, size_t ws_size,
                              hipStream_t stream) {
  const float* x     = (const float*)d_in[0];   // [8, 2500, 100]
  const float* label = (const float*)d_in[1];   // [8922, 100]
  float* out = (float*)d_out;                   // [8, 8922, 100]
  (void)in_sizes; (void)n_in; (void)out_size; (void)d_ws; (void)ws_size;
  dim3 grid((NQT + NWAVE - 1) / NWAVE, BATCH);  // (70, 8)
  dim3 block(NTHR);                             // 8 waves
  hipLaunchKernelGGL(cwa_flash_f16wmma, grid, block, 0, stream, x, label, out);
}
